// FCOS_64072322122238
// MI455X (gfx1250) — compile-verified, matrix operand order checked by reading, not run on hardware
//
#include <hip/hip_runtime.h>

// ---------------------------------------------------------------------------
// Types for CDNA5 WMMA (wave32): D(f32 16x16) = A(f16 16x32) x B(f16 32x16) + C
// ---------------------------------------------------------------------------
typedef __attribute__((ext_vector_type(8)))  _Float16 v8h;
typedef __attribute__((ext_vector_type(16))) _Float16 v16h;
typedef __attribute__((ext_vector_type(8)))  float    v8f;
typedef __attribute__((ext_vector_type(4)))  int      v4i;

#define DEV __device__ __forceinline__

// ---------------------------------------------------------------------------
// CDNA5 async global->LDS staging (ASYNCcnt-tracked), with fallback.
// Builtin signature (probe-verified): (int4 AS1*, int4 AS3*, imm off, imm cpol)
// ---------------------------------------------------------------------------
#if __has_builtin(__builtin_amdgcn_global_load_async_to_lds_b128)
#define USE_ASYNC_LDS 1
#else
#define USE_ASYNC_LDS 0
#endif

#if USE_ASYNC_LDS
typedef __attribute__((address_space(1))) v4i as1_v4i;
typedef __attribute__((address_space(3))) v4i as3_v4i;

template <int OFF>
DEV void async_copy_b128(const void* g, void* l) {
  __builtin_amdgcn_global_load_async_to_lds_b128(
      (as1_v4i*)const_cast<void*>(g), (as3_v4i*)l, OFF, 0);
}

DEV void wait_async() {
#if __has_builtin(__builtin_amdgcn_s_wait_asynccnt)
  __builtin_amdgcn_s_wait_asynccnt(0);
#else
  asm volatile("s_wait_asynccnt 0x0" ::: "memory");
#endif
}
#endif

// Fragment load from LDS: one matrix row (stride-40-half padded) -> v16h.
// Lanes 0-15 take K=[0..7]+[16..23], lanes 16-31 take K=[8..15]+[24..31].
DEV v16h load_frag(const _Float16* row, int klo) {
  v8h lo = *(const v8h*)(row + klo);
  v8h hi = *(const v8h*)(row + klo + 16);
  v16h r;
#pragma unroll
  for (int i = 0; i < 8; ++i) { r[i] = lo[i]; r[i + 8] = hi[i]; }
  return r;
}

DEV v8f wmma16(v16h a, v16h b, v8f c) {
  return __builtin_amdgcn_wmma_f32_16x16x32_f16(false, a, false, b, (short)0, c,
                                                false, false);
}

// ---------------------------------------------------------------------------
// Weight packing: fp32 [L][co][ci][3][3] -> fp16 [L][r][co][ci], r = kh*3+kw
// ---------------------------------------------------------------------------
__global__ void pack_stem_k(const float* __restrict__ w, _Float16* __restrict__ wp,
                            int total) {
  int i = blockIdx.x * 256 + threadIdx.x;
  if (i >= total) return;
  int ci = i & 255;
  int t = i >> 8;
  int co = t & 255;
  t >>= 8;               // t = l*9 + r
  int r = t % 9;
  int l = t / 9;
  wp[i] = (_Float16)w[(((size_t)l * 256 + co) * 256 + ci) * 9 + r];
}

__global__ void pack_head_k(const float* __restrict__ w, _Float16* __restrict__ wp,
                            int coN, int Mpad) {
  int i = blockIdx.x * 256 + threadIdx.x;
  int total = 9 * Mpad * 256;
  if (i >= total) return;
  int ci = i & 255;
  int t = i >> 8;
  int co = t % Mpad;
  int r = t / Mpad;
  wp[i] = (co < coN) ? (_Float16)w[((size_t)co * 256 + ci) * 9 + r] : (_Float16)0.f;
}

__global__ void pack_boxctr_k(const float* __restrict__ wb, const float* __restrict__ wc,
                              _Float16* __restrict__ wp) {
  int i = blockIdx.x * 256 + threadIdx.x;
  if (i >= 9 * 16 * 256) return;
  int ci = i & 255;
  int t = i >> 8;
  int co = t & 15;
  int r = t >> 4;
  float v = 0.f;
  if (co < 4)       v = wb[((size_t)co * 256 + ci) * 9 + r];
  else if (co == 4) v = wc[(size_t)ci * 9 + r];
  wp[i] = (_Float16)v;
}

__global__ void make_bias16_k(const float* __restrict__ bb, const float* __restrict__ cb,
                              float* __restrict__ out) {
  int i = threadIdx.x;  // 16 threads
  out[i] = (i < 4) ? bb[i] : ((i == 4) ? cb[0] : 0.f);
}

// ---------------------------------------------------------------------------
// NCHW fp32 -> channels-last [n][p][c] fp16, LDS-tiled 32x32 transpose.
// ---------------------------------------------------------------------------
__global__ __launch_bounds__(256) void nchw_to_pl_k(
    const float* __restrict__ src, _Float16* __restrict__ dst, int HW) {
  __shared__ _Float16 t[32][33];
  const int n  = blockIdx.z;
  const int pB = blockIdx.x * 32;
  const int cB = blockIdx.y * 32;
  const int tx = threadIdx.x & 31;
  const int ty = threadIdx.x >> 5;  // 0..7
#pragma unroll
  for (int j = 0; j < 32; j += 8) {
    const int c = cB + ty + j;
    t[ty + j][tx] = (_Float16)src[((size_t)n * 256 + c) * HW + pB + tx];
  }
  __syncthreads();
#pragma unroll
  for (int j = 0; j < 32; j += 8) {
    const int pp = pB + ty + j;
    dst[((size_t)n * HW + pp) * 256 + cB + tx] = t[tx][ty + j];
  }
}

// ---------------------------------------------------------------------------
// Stem conv staging helpers: issue (async DMA or reg loads), commit (wait or
// ds_store). Each thread stages 32B of A and 32B of B per K-step.
// ---------------------------------------------------------------------------
DEV void stage_issue(const _Float16* aRow, const _Float16* bRow, bool inb, int kb,
                     _Float16* Asl, _Float16* Bsl, int srow, int skof, v8h rg[4]) {
  const _Float16* ga = aRow + kb;
  const _Float16* gb = bRow + kb;
  _Float16* la = Asl + srow * 40 + skof;
  _Float16* lb = Bsl + srow * 40 + skof;
#if USE_ASYNC_LDS
  (void)rg;
  async_copy_b128<0>(ga, la);
  async_copy_b128<16>(ga, la);
  if (inb) {
    async_copy_b128<0>(gb, lb);
    async_copy_b128<16>(gb, lb);
  } else {
    v8h z;
#pragma unroll
    for (int j = 0; j < 8; ++j) z[j] = (_Float16)0.f;
    *(v8h*)lb = z;
    *(v8h*)(lb + 8) = z;
  }
#else
  (void)la; (void)lb;
  rg[0] = *(const v8h*)ga;
  rg[1] = *(const v8h*)(ga + 8);
  if (inb) {
    rg[2] = *(const v8h*)gb;
    rg[3] = *(const v8h*)(gb + 8);
  } else {
#pragma unroll
    for (int j = 0; j < 8; ++j) { rg[2][j] = (_Float16)0.f; rg[3][j] = (_Float16)0.f; }
  }
#endif
}

DEV void stage_commit(_Float16* Asl, _Float16* Bsl, int srow, int skof,
                      const v8h rg[4]) {
#if USE_ASYNC_LDS
  (void)Asl; (void)Bsl; (void)srow; (void)skof; (void)rg;
  wait_async();
#else
  *(v8h*)(Asl + srow * 40 + skof)     = rg[0];
  *(v8h*)(Asl + srow * 40 + skof + 8) = rg[1];
  *(v8h*)(Bsl + srow * 40 + skof)     = rg[2];
  *(v8h*)(Bsl + srow * 40 + skof + 8) = rg[3];
#endif
}

// ---------------------------------------------------------------------------
// Stem conv: 3x3 SAME, 256->256, bias+ReLU, fp16 channels-last in/out.
// Implicit GEMM: M = out-ch, N = positions, K = r*256+ci (9 taps x 8 blocks).
// Workgroup: 256 thr (8 waves), tile 128(M) x 128(N); wave = 32x64 (2x4 frags).
// Double-buffered LDS, one barrier per K-step, async DMA staging of step s+1
// overlapped with the 8 WMMAs of step s. All 12 fragment ds_loads are hoisted
// ahead of the WMMA burst so dscnt can drain incrementally.
// ---------------------------------------------------------------------------
__global__ __launch_bounds__(256) void stem_conv_k(
    const _Float16* __restrict__ X,   // [16][HW][256] channels-last
    const _Float16* __restrict__ Wp,  // [9][256][256]
    const float* __restrict__ bias,   // [256]
    _Float16* __restrict__ Y,         // [16][HW][256]
    int H, int W) {
  const int HW = H * W;
  const int n      = blockIdx.z;
  const int coBase = blockIdx.y * 128;
  const int pBase  = blockIdx.x * 128;

  const int tid  = threadIdx.x;
  const int lane = tid & 31;
  const int wave = tid >> 5;
  const int wm   = wave >> 1;  // 0..3 : M sub-tile (32 rows)
  const int wn   = wave & 1;   // 0..1 : N sub-tile (64 cols)

  __shared__ _Float16 As[2][128 * 40];
  __shared__ _Float16 Bs[2][128 * 40];

  v8f acc[2][4] = {};

  // staging coords: thread -> (row/pos, 16-half k group)
  const int srow = tid >> 1;        // 0..127
  const int skof = (tid & 1) * 16;  // 0 | 16
  const int p  = pBase + srow;      // HW divisible by 128 at all levels
  const int ph = p / W, pw = p % W;
  const _Float16* aBase = Wp + (size_t)(coBase + srow) * 256 + skof;
  const _Float16* xBase = X + (size_t)n * HW * 256 + skof;

  const int klo = (lane < 16) ? 0 : 8;
  const int lr  = lane & 15;

  // B-tile coordinates for a given tap r (hoisted per-r, not per-step)
  auto calcB = [&](int r, const _Float16*& bRow, bool& inb) {
    const int hh = ph + r / 3 - 1;
    const int ww = pw + r % 3 - 1;
    inb = (hh >= 0) && (hh < H) && (ww >= 0) && (ww < W);
    bRow = xBase + (size_t)(hh * W + ww) * 256;
  };

  const _Float16* bRow; bool inb;
  calcB(0, bRow, inb);
  const _Float16* aRow = aBase;
  v8h rg[4];

  stage_issue(aRow, bRow, inb, 0, As[0], Bs[0], srow, skof, rg);
  stage_commit(As[0], Bs[0], srow, skof, rg);
  __syncthreads();

  for (int r = 0; r < 9; ++r) {
    const _Float16* bRowN = bRow; bool inbN = inb;
    if (r < 8) calcB(r + 1, bRowN, inbN);
    const _Float16* aRowN = aBase + (size_t)(r + 1) * 65536;
#pragma unroll
    for (int j = 0; j < 8; ++j) {   // s = r*8+j; buffer parity = j&1 (static)
      const int cur = j & 1;
      const int nxt = cur ^ 1;
      const bool more = (r < 8) || (j < 7);
      if (more) {
        if (j < 7)
          stage_issue(aRow, bRow, inb, (j + 1) * 32, As[nxt], Bs[nxt], srow, skof, rg);
        else
          stage_issue(aRowN, bRowN, inbN, 0, As[nxt], Bs[nxt], srow, skof, rg);
      }

      // ---- hoist all 12 fragment loads, then burst 8 WMMAs ----
      const _Float16* Ab = As[cur];
      const _Float16* Bb = Bs[cur];
      v16h a0 = load_frag(Ab + (wm * 32 +      lr) * 40, klo);
      v16h a1 = load_frag(Ab + (wm * 32 + 16 + lr) * 40, klo);
      v16h b0 = load_frag(Bb + (wn * 64 +  0 + lr) * 40, klo);
      v16h b1 = load_frag(Bb + (wn * 64 + 16 + lr) * 40, klo);
      v16h b2 = load_frag(Bb + (wn * 64 + 32 + lr) * 40, klo);
      v16h b3 = load_frag(Bb + (wn * 64 + 48 + lr) * 40, klo);

      acc[0][0] = wmma16(a0, b0, acc[0][0]);
      acc[1][0] = wmma16(a1, b0, acc[1][0]);
      acc[0][1] = wmma16(a0, b1, acc[0][1]);
      acc[1][1] = wmma16(a1, b1, acc[1][1]);
      acc[0][2] = wmma16(a0, b2, acc[0][2]);
      acc[1][2] = wmma16(a1, b2, acc[1][2]);
      acc[0][3] = wmma16(a0, b3, acc[0][3]);
      acc[1][3] = wmma16(a1, b3, acc[1][3]);

      if (more) stage_commit(As[nxt], Bs[nxt], srow, skof, rg);
      __syncthreads();
    }
    bRow = bRowN; inb = inbN; aRow = aRowN;
  }

  // ---- store: bias + ReLU, fp16 channels-last, one b128 per fragment ----
  const int mofs = (lane < 16) ? 0 : 8;
#pragma unroll
  for (int mi = 0; mi < 2; ++mi) {
#pragma unroll
    for (int ni = 0; ni < 4; ++ni) {
      const int co0 = coBase + wm * 32 + mi * 16 + mofs;
      const int pT  = pBase + wn * 64 + ni * 16 + lr;
      _Float16* yb = Y + ((size_t)(n * HW + pT)) * 256 + co0;
      v8h o;
#pragma unroll
      for (int v = 0; v < 8; ++v) {
        const float val = acc[mi][ni][v] + bias[co0 + v];
        o[v] = (_Float16)fmaxf(val, 0.f);
      }
      *(v8h*)yb = o;
    }
  }
}

// ---------------------------------------------------------------------------
// Pred conv: 3x3 SAME, 256->Mpad (<=80), bias, fp32 channel-last output written
// directly into the (16, 5376, 85) result. Workgroup: 16(M) x 128(N), 8 waves.
// ---------------------------------------------------------------------------
__global__ __launch_bounds__(256) void pred_conv_k(
    const _Float16* __restrict__ X,   // [16][HW][256] channels-last
    const _Float16* __restrict__ Wp,  // [9][Mpad][256]
    const float* __restrict__ bias,   // [Mpad]
    float* __restrict__ Out,          // [16][5376][85]
    int H, int W, int Mpad, int coCount, int chOff, int levelOff) {
  const int HW = H * W;
  const int n      = blockIdx.z;
  const int coBase = blockIdx.y * 16;
  const int pBase  = blockIdx.x * 128;

  const int tid  = threadIdx.x;
  const int lane = tid & 31;
  const int wave = tid >> 5;  // 0..7 : N sub-tile

  __shared__ _Float16 As[16 * 40];
  __shared__ _Float16 Bs[128 * 40];

  v8f acc = {};

  const int arow = tid >> 4;        // 0..15
  const int ak2  = (tid & 15) * 2;  // 0..30 step 2
  const int bp   = tid >> 1;        // 0..127
  const int bk   = (tid & 1) * 16;  // 0 | 16
  const int p  = pBase + bp;
  const int ph = p / W, pw = p % W;

  const int klo = (lane < 16) ? 0 : 8;
  const int lr  = lane & 15;

  for (int r = 0; r < 9; ++r) {
    const int dh = r / 3 - 1, dw = r % 3 - 1;
    const _Float16* Wr = Wp + (size_t)r * Mpad * 256;
    const int hh = ph + dh, ww = pw + dw;
    const bool inb = (hh >= 0) && (hh < H) && (ww >= 0) && (ww < W);
    const int sp = hh * W + ww;

    for (int kb = 0; kb < 256; kb += 32) {
      *(unsigned*)(As + arow * 40 + ak2) =
          *(const unsigned*)(Wr + (size_t)(coBase + arow) * 256 + kb + ak2);
      {
        v8h b0, b1;
        if (inb) {
          const _Float16* bsrc = X + ((size_t)(n * HW + sp)) * 256 + kb + bk;
          b0 = *(const v8h*)bsrc;
          b1 = *(const v8h*)(bsrc + 8);
        } else {
#pragma unroll
          for (int j = 0; j < 8; ++j) { b0[j] = (_Float16)0.f; b1[j] = (_Float16)0.f; }
        }
        *(v8h*)(Bs + bp * 40 + bk)     = b0;
        *(v8h*)(Bs + bp * 40 + bk + 8) = b1;
      }
      __syncthreads();

      v16h a = load_frag(As + lr * 40, klo);
      v16h b = load_frag(Bs + (wave * 16 + lr) * 40, klo);
      acc = wmma16(a, b, acc);
      __syncthreads();
    }
  }

  const int mofs = (lane < 16) ? 0 : 8;
  const int pT   = pBase + wave * 16 + lr;
  float* ob = Out + ((size_t)n * 5376 + levelOff + pT) * 85 + chOff;
#pragma unroll
  for (int v = 0; v < 8; ++v) {
    const int co = coBase + mofs + v;
    if (co < coCount) ob[co] = acc[v] + bias[co];
  }
}

// ---------------------------------------------------------------------------
// Host orchestration
// ---------------------------------------------------------------------------
extern "C" void kernel_launch(void* const* d_in, const int* in_sizes, int n_in,
                              void* d_out, int out_size, void* d_ws, size_t ws_size,
                              hipStream_t stream) {
  (void)in_sizes; (void)n_in; (void)out_size; (void)ws_size;

  const float* feat[3]   = {(const float*)d_in[0], (const float*)d_in[1],
                            (const float*)d_in[2]};
  const float* stem_w[2] = {(const float*)d_in[3], (const float*)d_in[5]};  // cls, box
  const float* stem_b[2] = {(const float*)d_in[4], (const float*)d_in[6]};
  const float* pcw = (const float*)d_in[7];   // pred_cls_w (80,256,3,3)
  const float* pcb = (const float*)d_in[8];   // pred_cls_b (80)
  const float* pbw = (const float*)d_in[9];   // pred_box_w (4,256,3,3)
  const float* pbb = (const float*)d_in[10];  // pred_box_b (4)
  const float* pww = (const float*)d_in[11];  // pred_ctr_w (1,256,3,3)
  const float* pwb = (const float*)d_in[12];  // pred_ctr_b (1)
  float* out = (float*)d_out;

  // ---- workspace layout (fp16) ----
  _Float16* ws16 = (_Float16*)d_ws;
  const size_t STEMW = (size_t)4 * 9 * 256 * 256;
  _Float16* wst[2] = {ws16, ws16 + STEMW};
  _Float16* wpc = ws16 + 2 * STEMW;                 // [9][80][256]
  _Float16* wpb = wpc + (size_t)9 * 80 * 256;       // [9][16][256]
  float* bias16 = (float*)(wpb + (size_t)9 * 16 * 256);
  _Float16* act[2];
  act[0] = (_Float16*)(bias16 + 16);
  act[1] = act[0] + (size_t)16 * 4096 * 256;        // sized for P3, channels-last

  // ---- pack weights once per call ----
  {
    const int total = 4 * 9 * 256 * 256;
    const int blocks = (total + 255) / 256;
    pack_stem_k<<<blocks, 256, 0, stream>>>(stem_w[0], wst[0], total);
    pack_stem_k<<<blocks, 256, 0, stream>>>(stem_w[1], wst[1], total);
    pack_head_k<<<(9 * 80 * 256 + 255) / 256, 256, 0, stream>>>(pcw, wpc, 80, 80);
    pack_boxctr_k<<<(9 * 16 * 256 + 255) / 256, 256, 0, stream>>>(pbw, pww, wpb);
    make_bias16_k<<<1, 16, 0, stream>>>(pbb, pwb, bias16);
  }

  const int Hs[3]   = {64, 32, 16};
  const int offs[3] = {0, 4096, 5120};

  for (int L = 0; L < 3; ++L) {
    const int H = Hs[L], W = Hs[L], HW = H * W;
    for (int br = 0; br < 2; ++br) {  // 0 = cls branch, 1 = box branch
      {
        dim3 g(HW / 32, 8, 16);
        nchw_to_pl_k<<<g, 256, 0, stream>>>(feat[L], act[0], HW);
      }
      int cur = 0;
      for (int l = 0; l < 4; ++l) {
        dim3 g(HW / 128, 2, 16);
        stem_conv_k<<<g, 256, 0, stream>>>(act[cur],
                                           wst[br] + (size_t)l * 9 * 256 * 256,
                                           stem_b[br] + l * 256, act[cur ^ 1], H, W);
        cur ^= 1;
      }
      // cur == 0 again after 4 layers
      if (br == 0) {
        dim3 g(HW / 128, 5, 16);
        pred_conv_k<<<g, 256, 0, stream>>>(act[cur], wpc, pcb, out, H, W,
                                           80, 80, 0, offs[L]);
      } else {
        dim3 g(HW / 128, 1, 16);
        pred_conv_k<<<g, 256, 0, stream>>>(act[cur], wpb, bias16, out, H, W,
                                           16, 5, 80, offs[L]);
      }
    }
  }
}
